// LASGlobal_53360673686037
// MI455X (gfx1250) — matrix operand order repaired and checked by reading, not compiled, not run on hardware
//
#include <hip/hip_runtime.h>
#include <hip/hip_bf16.h>

// ---------------- types / WMMA helpers ----------------
typedef __attribute__((ext_vector_type(16))) __bf16 v16bf;
typedef __attribute__((ext_vector_type(8)))  float  v8f;
typedef __attribute__((ext_vector_type(4)))  unsigned int u32x4;

#define DEV static __device__ __forceinline__

DEV v8f wmma_bf16(v16bf a, v16bf b, v8f c) {
  return __builtin_amdgcn_wmma_f32_16x16x32_bf16(false, a, false, b, (short)0, c, false, false);
}

union FragU { v16bf v; u32x4 q[2]; };

// A fragment (16x32, bf16), row-major source with row already selected.
// ISA layout: lanes 0-15 hold K={kb+0..7, kb+16..23}; lanes 16-31 K={kb+8..15, kb+24..31}.
DEV v16bf load_a_frag(const __bf16* rowbase, int kb, int lane) {
  FragU u;
  const __bf16* p = rowbase + kb + ((lane >> 4) << 3);
  u.q[0] = *(const u32x4*)(p);
  u.q[1] = *(const u32x4*)(p + 16);
  return u.v;
}
// B fragment (32x16) from W stored (N,K) row-major (i.e. computes A @ W^T).
// lanes 0-15: N=n, K={kb+0..7, kb+16..23}; lanes 16-31: K={kb+8..15, kb+24..31}.
DEV v16bf load_b_frag(const __bf16* W, long K, int n, int kb, int lane) {
  FragU u;
  const __bf16* p = W + (long)n * K + kb + ((lane >> 4) << 3);
  u.q[0] = *(const u32x4*)(p);
  u.q[1] = *(const u32x4*)(p + 16);
  return u.v;
}

DEV float sigf(float x) { return 1.f / (1.f + __expf(-x)); }

// ---------------- model dims ----------------
constexpr int BB   = 8;
constexpr int TT   = 1600;
constexpr int FF   = 80;
constexpr int TE   = 400;     // TT / SUB
constexpr int EH   = 512;
constexpr int G4   = 2048;    // 4*EH
constexpr int EOUT = 1024;    // 2*EH
constexpr int UU   = 100;
constexpr int VV   = 5000;
constexpr int DCAT = 1536;    // DH + EOUT

// ---------------- utility kernels ----------------
__global__ void k_cvt_bf16(const float* __restrict__ s, __bf16* __restrict__ d, int n) {
  int i = blockIdx.x * 256 + threadIdx.x;
  if (i < n) d[i] = (__bf16)s[i];
}

// feats (8,1600,80) -> bf16 (8,400,320) frame-stacked
__global__ void k_subsample(const float* __restrict__ feats, __bf16* __restrict__ x, int n) {
  int i = blockIdx.x * 256 + threadIdx.x;
  if (i >= n) return;
  int j = i % 320;
  int te = (i / 320) % TE;
  int b = i / (320 * TE);
  int f = j % FF, f4 = j / FF;
  x[i] = (__bf16)feats[((long)b * TT + te * 4 + f4) * FF + f];
}

// ---------------- generic WMMA GEMM: C[M,N] = A[M,K] @ W[N,K]^T (+bias) ----------------
// block = 128 threads (4 waves). Wave -> one N-tile x four M-tiles (B-frag reused 4x).
// grid = (ceil(M/64), ceil(N/64)).
__global__ __launch_bounds__(128)
void k_gemm_bf16(const __bf16* __restrict__ A, const __bf16* __restrict__ W,
                 const float* __restrict__ bias, float* __restrict__ C,
                 int M, int N, int K) {
  const int w = threadIdx.x >> 5, l = threadIdx.x & 31;
  const int tn = blockIdx.y * 4 + w;
  int nload = tn * 16 + (l & 15);
  if (nload >= N) nload = N - 1;          // clamp (stores are guarded)
  const __bf16* Arow[4];
#pragma unroll
  for (int mt = 0; mt < 4; ++mt) {
    int m = (blockIdx.x * 4 + mt) * 16 + (l & 15);
    if (m >= M) m = M - 1;                // clamp (stores are guarded)
    Arow[mt] = A + (long)m * K;
  }
  v8f z = {};
  v8f acc[4];
#pragma unroll
  for (int mt = 0; mt < 4; ++mt) acc[mt] = z;

  for (int kb = 0; kb < K; kb += 32) {
    v16bf b = load_b_frag(W, K, nload, kb, l);
#pragma unroll
    for (int mt = 0; mt < 4; ++mt) {
      v16bf a = load_a_frag(Arow[mt], kb, l);
      acc[mt] = wmma_bf16(a, b, acc[mt]);
    }
  }
  int n = tn * 16 + (l & 15);
  if (n < N) {
    float bv = bias ? bias[n] : 0.f;
#pragma unroll
    for (int mt = 0; mt < 4; ++mt) {
      int mbase = (blockIdx.x * 4 + mt) * 16 + ((l >> 4) << 3);
#pragma unroll
      for (int r = 0; r < 8; ++r) {
        int m = mbase + r;
        if (m < M) C[(long)m * N + n] = acc[mt][r] + bv;
      }
    }
  }
}

// ---------------- persistent bidirectional LSTM layer step kernel ----------------
// grid.x = 2 (dir). 512 threads = 16 waves. h (bf16) + c (f32) resident in LDS.
// Wave w owns gate columns [w*32, w*32+32) of all 4 gates -> 8 accumulators.
__global__ __launch_bounds__(512)
void k_enc_rnn(const float* __restrict__ xW_f, const float* __restrict__ xW_b,
               const __bf16* __restrict__ Whh_f, const __bf16* __restrict__ Whh_b,
               const float* __restrict__ bih_f, const float* __restrict__ bhh_f,
               const float* __restrict__ bih_b, const float* __restrict__ bhh_b,
               const int* __restrict__ feat_lens, float* __restrict__ out) {
  __shared__ __bf16 hls[16 * EH];
  __shared__ float  cls[16 * EH];
  const int dir = blockIdx.x;
  const float*  xW  = dir ? xW_b : xW_f;
  const __bf16* Whh = dir ? Whh_b : Whh_f;
  const float*  bih = dir ? bih_b : bih_f;
  const float*  bhh = dir ? bhh_b : bhh_f;
  const int tid = threadIdx.x, w = tid >> 5, l = tid & 31;
  const int mrow = (l >> 4) << 3;
  for (int i = tid; i < 16 * EH; i += 512) { hls[i] = (__bf16)0.f; cls[i] = 0.f; }
  __syncthreads();

  for (int s = 0; s < TE; ++s) {
    v8f z = {};
    v8f acc[4][2];
#pragma unroll
    for (int g = 0; g < 4; ++g) { acc[g][0] = z; acc[g][1] = z; }
    const __bf16* Arow = hls + (long)(l & 15) * EH;
    for (int kb = 0; kb < EH; kb += 32) {
      v16bf a = load_a_frag(Arow, kb, l);
#pragma unroll
      for (int g = 0; g < 4; ++g)
#pragma unroll
        for (int jt = 0; jt < 2; ++jt) {
          int n = g * EH + w * 32 + jt * 16 + (l & 15);
          v16bf b = load_b_frag(Whh, EH, n, kb, l);
          acc[g][jt] = wmma_bf16(a, b, acc[g][jt]);
        }
    }
    __syncthreads();   // all waves done reading hls
#pragma unroll
    for (int jt = 0; jt < 2; ++jt) {
      int nj = w * 32 + jt * 16 + (l & 15);   // 0..511
#pragma unroll
      for (int r = 0; r < 8; ++r) {
        int m = mrow + r;
        if (m >= BB) continue;                // padding rows stay zero
        int len = (feat_lens[m] + 3) >> 2;
        bool valid = s < len;
        int tx = dir ? (valid ? len - 1 - s : s) : s;
        long xwb = ((long)m * TE + tx) * G4;
        float gi = acc[0][jt][r] + xW[xwb + nj]            + bih[nj]            + bhh[nj];
        float gf = acc[1][jt][r] + xW[xwb + EH + nj]       + bih[EH + nj]       + bhh[EH + nj];
        float gg = acc[2][jt][r] + xW[xwb + 2 * EH + nj]   + bih[2 * EH + nj]   + bhh[2 * EH + nj];
        float go = acc[3][jt][r] + xW[xwb + 3 * EH + nj]   + bih[3 * EH + nj]   + bhh[3 * EH + nj];
        float c2 = sigf(gf) * cls[m * EH + nj] + sigf(gi) * tanhf(gg);
        float h2 = sigf(go) * tanhf(c2);
        int pos; float hv;
        if (valid) {
          cls[m * EH + nj] = c2;
          hls[m * EH + nj] = (__bf16)h2;
          pos = dir ? (len - 1 - s) : s; hv = h2;
        } else { pos = s; hv = 0.f; }
        out[((long)m * TE + pos) * EOUT + dir * EH + nj] = hv;
      }
    }
    __syncthreads();   // hls updated for next step
  }
}

// ---------------- persistent attention decoder ----------------
// 1 block, 512 threads (16 waves). Abuf (bf16 16x2048) = [emb | ctx | h] is the WMMA A-matrix.
__global__ __launch_bounds__(512)
void k_decoder(const float* __restrict__ emb, const int* __restrict__ y_in,
               const __bf16* __restrict__ Wih, const __bf16* __restrict__ Whh,
               const float* __restrict__ bih, const float* __restrict__ bhh,
               const __bf16* __restrict__ Wsb, const float* __restrict__ vvec,
               const float* __restrict__ hproj, const float* __restrict__ enc_out,
               const int* __restrict__ feat_lens, __bf16* __restrict__ dec_cat) {
  extern __shared__ char smem[];
  __bf16* Abuf = (__bf16*)smem;                         // 16 x 2048
  float*  cls  = (float*)(Abuf + 16 * 2048);            // 16 x 512
  float*  sls  = cls + 16 * EH;                         // 16 x 512
  float*  els  = sls + 16 * EH;                         // 8 x 400
  const int tid = threadIdx.x, w = tid >> 5, l = tid & 31;
  const int mrow = (l >> 4) << 3;
  for (int i = tid; i < 16 * 2048; i += 512) Abuf[i] = (__bf16)0.f;
  for (int i = tid; i < 16 * EH; i += 512) cls[i] = 0.f;
  __syncthreads();

  for (int t = 0; t < UU; ++t) {
    // 1. embedding gather -> Abuf[:,0:512)
    for (int i = tid; i < BB * 512; i += 512) {
      int b = i >> 9, d = i & 511;
      Abuf[b * 2048 + d] = (__bf16)emb[(long)y_in[b * UU + t] * 512 + d];
    }
    __syncthreads();

    // 2. gates = [emb,ctx] @ Wih^T + h @ Whh^T  (K = 1536 then 512)
    v8f z = {};
    v8f acc[4][2];
#pragma unroll
    for (int g = 0; g < 4; ++g) { acc[g][0] = z; acc[g][1] = z; }
    const __bf16* Arow = Abuf + (long)(l & 15) * 2048;
    for (int kb = 0; kb < 1536; kb += 32) {
      v16bf a = load_a_frag(Arow, kb, l);
#pragma unroll
      for (int g = 0; g < 4; ++g)
#pragma unroll
        for (int jt = 0; jt < 2; ++jt) {
          int n = g * EH + w * 32 + jt * 16 + (l & 15);
          v16bf b = load_b_frag(Wih, 1536, n, kb, l);
          acc[g][jt] = wmma_bf16(a, b, acc[g][jt]);
        }
    }
    for (int kb = 0; kb < EH; kb += 32) {
      v16bf a = load_a_frag(Arow + 1536, kb, l);
#pragma unroll
      for (int g = 0; g < 4; ++g)
#pragma unroll
        for (int jt = 0; jt < 2; ++jt) {
          int n = g * EH + w * 32 + jt * 16 + (l & 15);
          v16bf b = load_b_frag(Whh, EH, n, kb, l);
          acc[g][jt] = wmma_bf16(a, b, acc[g][jt]);
        }
    }
    __syncthreads();

    // 3. LSTM elementwise -> h2 (bf16 into Abuf h-slot + dec_cat), c2
#pragma unroll
    for (int jt = 0; jt < 2; ++jt) {
      int nj = w * 32 + jt * 16 + (l & 15);
#pragma unroll
      for (int r = 0; r < 8; ++r) {
        int m = mrow + r;
        if (m >= BB) continue;
        float gi = acc[0][jt][r] + bih[nj]          + bhh[nj];
        float gf = acc[1][jt][r] + bih[EH + nj]     + bhh[EH + nj];
        float gg = acc[2][jt][r] + bih[2 * EH + nj] + bhh[2 * EH + nj];
        float go = acc[3][jt][r] + bih[3 * EH + nj] + bhh[3 * EH + nj];
        float c2 = sigf(gf) * cls[m * EH + nj] + sigf(gi) * tanhf(gg);
        float h2 = sigf(go) * tanhf(c2);
        cls[m * EH + nj] = c2;
        __bf16 hb = (__bf16)h2;
        Abuf[m * 2048 + 1536 + nj] = hb;
        dec_cat[((long)m * UU + t) * DCAT + nj] = hb;
      }
    }
    __syncthreads();

    // 4. s = h2 @ Ws^T  (16 x 512, K=512)
    v8f sacc[2]; sacc[0] = z; sacc[1] = z;
    const __bf16* Hrow = Abuf + (long)(l & 15) * 2048 + 1536;
    for (int kb = 0; kb < EH; kb += 32) {
      v16bf a = load_a_frag(Hrow, kb, l);
#pragma unroll
      for (int jt = 0; jt < 2; ++jt) {
        int n = w * 32 + jt * 16 + (l & 15);
        v16bf b = load_b_frag(Wsb, EH, n, kb, l);
        sacc[jt] = wmma_bf16(a, b, sacc[jt]);
      }
    }
#pragma unroll
    for (int jt = 0; jt < 2; ++jt) {
      int n = w * 32 + jt * 16 + (l & 15);
#pragma unroll
      for (int r = 0; r < 8; ++r) sls[(mrow + r) * EH + n] = sacc[jt][r];
    }
    __syncthreads();

    // 5. energies e[b,te] = tanh(hproj + s) . v
    for (int idx = tid; idx < BB * TE; idx += 512) {
      int b = idx / TE, te = idx % TE;
      const float* hp = hproj + ((long)b * TE + te) * 512;
      const float* sb = sls + b * EH;
      float e = 0.f;
      for (int k = 0; k < 512; ++k) e += tanhf(hp[k] + sb[k]) * vvec[k];
      els[idx] = e;
    }
    __syncthreads();

    // 6. masked softmax, one wave32 per batch row
    if (w < BB) {
      int len = (feat_lens[w] + 3) >> 2;
      float mx = -3.4e38f;
      for (int te = l; te < TE; te += 32) if (te < len) mx = fmaxf(mx, els[w * TE + te]);
      for (int o = 16; o > 0; o >>= 1) mx = fmaxf(mx, __shfl_xor(mx, o, 32));
      float sum = 0.f;
      for (int te = l; te < TE; te += 32) {
        float aa = (te < len) ? __expf(els[w * TE + te] - mx) : 0.f;
        els[w * TE + te] = aa; sum += aa;
      }
      for (int o = 16; o > 0; o >>= 1) sum += __shfl_xor(sum, o, 32);
      float inv = 1.f / sum;
      for (int te = l; te < TE; te += 32) els[w * TE + te] *= inv;
    }
    __syncthreads();

    // 7. ctx = a @ enc_out -> Abuf ctx-slot + dec_cat
    for (int idx = tid; idx < BB * EOUT; idx += 512) {
      int b = idx >> 10, d = idx & 1023;
      const float* eo = enc_out + (long)b * TE * EOUT + d;
      const float* ab = els + b * TE;
      float cv = 0.f;
      for (int te = 0; te < TE; ++te) cv += ab[te] * eo[(long)te * EOUT];
      __bf16 cb = (__bf16)cv;
      Abuf[b * 2048 + 512 + d] = cb;
      dec_cat[((long)b * UU + t) * DCAT + 512 + d] = cb;
    }
    __syncthreads();
  }
}

// ---------------- host orchestration ----------------
extern "C" void kernel_launch(void* const* d_in, const int* in_sizes, int n_in,
                              void* d_out, int out_size, void* d_ws, size_t ws_size,
                              hipStream_t stream) {
  (void)in_sizes; (void)n_in; (void)out_size; (void)ws_size;
  const float* feats     = (const float*)d_in[0];
  const int*   feat_lens = (const int*)d_in[1];
  const int*   y_in      = (const int*)d_in[2];
  // enc_params: layer L base = 3 + L*8 : Wih_f,Whh_f,bih_f,bhh_f,Wih_b,Whh_b,bih_b,bhh_b
  const float* emb     = (const float*)d_in[27];
  const float* dec_Wih = (const float*)d_in[28];
  const float* dec_Whh = (const float*)d_in[29];
  const float* dec_bih = (const float*)d_in[30];
  const float* dec_bhh = (const float*)d_in[31];
  const float* Wh      = (const float*)d_in[32];
  const float* Ws      = (const float*)d_in[33];
  const float* vvec    = (const float*)d_in[34];
  const float* out_W   = (const float*)d_in[35];
  const float* out_b   = (const float*)d_in[36];

  char* base = (char*)d_ws; size_t off = 0;
  auto alloc = [&](size_t bytes) -> void* {
    void* r = base + off; off = (off + bytes + 255) & ~(size_t)255; return r;
  };
  const int MR = BB * TE; // 3200 GEMM rows

  __bf16* Xbf    = (__bf16*)alloc((size_t)MR * EOUT * 2);    // layer input (bf16)
  float*  Yfp    = (float*)alloc((size_t)MR * EOUT * 4);     // layer output / enc_out (f32)
  float*  xWf    = (float*)alloc((size_t)MR * G4 * 4);
  float*  xWb    = (float*)alloc((size_t)MR * G4 * 4);
  float*  hproj  = (float*)alloc((size_t)MR * EH * 4);
  __bf16* WihFb  = (__bf16*)alloc((size_t)G4 * EOUT * 2);
  __bf16* WihBb  = (__bf16*)alloc((size_t)G4 * EOUT * 2);
  __bf16* WhhFb  = (__bf16*)alloc((size_t)G4 * EH * 2);
  __bf16* WhhBb  = (__bf16*)alloc((size_t)G4 * EH * 2);
  __bf16* dWihb  = (__bf16*)alloc((size_t)G4 * DCAT * 2);
  __bf16* dWhhb  = (__bf16*)alloc((size_t)G4 * EH * 2);
  __bf16* Wsbb   = (__bf16*)alloc((size_t)EH * EH * 2);
  __bf16* Whbb   = (__bf16*)alloc((size_t)EH * EOUT * 2);
  __bf16* outWb  = (__bf16*)alloc((size_t)VV * DCAT * 2);
  __bf16* decCat = (__bf16*)alloc((size_t)BB * UU * DCAT * 2);

  auto cvt = [&](__bf16* dst, const float* src, int n) {
    k_cvt_bf16<<<(n + 255) / 256, 256, 0, stream>>>(src, dst, n);
  };

  // 0. subsample -> bf16
  {
    int n = MR * 320;
    k_subsample<<<(n + 255) / 256, 256, 0, stream>>>(feats, Xbf, n);
  }

  // 1. encoder layers
  int inw = 320;
  for (int L = 0; L < 3; ++L) {
    int pb = 3 + L * 8;
    const float* Wih_f = (const float*)d_in[pb + 0];
    const float* Whh_f = (const float*)d_in[pb + 1];
    const float* bih_f = (const float*)d_in[pb + 2];
    const float* bhh_f = (const float*)d_in[pb + 3];
    const float* Wih_b = (const float*)d_in[pb + 4];
    const float* Whh_b = (const float*)d_in[pb + 5];
    const float* bih_b = (const float*)d_in[pb + 6];
    const float* bhh_b = (const float*)d_in[pb + 7];
    cvt(WihFb, Wih_f, G4 * inw);  cvt(WhhFb, Whh_f, G4 * EH);
    cvt(WihBb, Wih_b, G4 * inw);  cvt(WhhBb, Whh_b, G4 * EH);
    dim3 g(MR / 64, G4 / 64);
    k_gemm_bf16<<<g, 128, 0, stream>>>(Xbf, WihFb, nullptr, xWf, MR, G4, inw);
    k_gemm_bf16<<<g, 128, 0, stream>>>(Xbf, WihBb, nullptr, xWb, MR, G4, inw);
    k_enc_rnn<<<2, 512, 0, stream>>>(xWf, xWb, WhhFb, WhhBb,
                                     bih_f, bhh_f, bih_b, bhh_b, feat_lens, Yfp);
    cvt(Xbf, Yfp, MR * EOUT);   // next layer input / final enc_out (bf16)
    inw = EOUT;
  }

  // 2. hproj = enc_out @ Wh^T
  cvt(Whbb, Wh, EH * EOUT);
  k_gemm_bf16<<<dim3(MR / 64, EH / 64), 128, 0, stream>>>(Xbf, Whbb, nullptr, hproj, MR, EH, EOUT);

  // 3. decoder (persistent, attention fused)
  cvt(dWihb, dec_Wih, G4 * DCAT);
  cvt(dWhhb, dec_Whh, G4 * EH);
  cvt(Wsbb, Ws, EH * EH);
  size_t smem = (size_t)16 * 2048 * 2 + 16 * EH * 4 * 2 + BB * TE * 4;
  k_decoder<<<1, 512, smem, stream>>>(emb, y_in, dWihb, dWhhb, dec_bih, dec_bhh,
                                      Wsbb, vvec, hproj, Yfp, feat_lens, decCat);

  // 4. logits = [h,ctx] @ out_W^T + out_b  (800 x 5000, K=1536)
  cvt(outWb, out_W, VV * DCAT);
  k_gemm_bf16<<<dim3((BB * UU + 63) / 64, (VV + 63) / 64), 128, 0, stream>>>(
      decCat, outWb, out_b, (float*)d_out, BB * UU, VV, DCAT);
}
